// CrossAttentionBlock_66898410602555
// MI455X (gfx1250) — matrix operand order repaired and checked by reading, not compiled, not run on hardware
//
#include <hip/hip_runtime.h>
#include <hip/hip_bf16.h>

typedef __attribute__((ext_vector_type(16))) _Float16 v16h;
typedef __attribute__((ext_vector_type(8)))  _Float16 v8h;
typedef __attribute__((ext_vector_type(8)))  float    v8f;
typedef __attribute__((ext_vector_type(4)))  unsigned int v4u;
typedef __attribute__((ext_vector_type(8)))  int      v8i;
typedef __attribute__((ext_vector_type(4)))  int      v4i;

#define DIM 768
#define NH  12
#define DH  64
#define PSEQ 2048
#define BATCH 2
#define MLPH 3072
#define ROWS (BATCH * PSEQ)   // 4096

__device__ __forceinline__ v16h pack16(v8h lo, v8h hi) {
    return __builtin_shufflevector(lo, hi, 0,1,2,3,4,5,6,7,8,9,10,11,12,13,14,15);
}

__device__ __forceinline__ v8f wmma_f16(v16h a, v16h b, v8f c) {
    return __builtin_amdgcn_wmma_f32_16x16x32_f16(false, a, false, b, (short)0, c, false, false);
}

// ---------------------------------------------------------------- TDM 2D tile load
// D# per CDNA5 ISA ch.8: group0 = {count/lds_addr/global_addr/type}, group1 packs
// data_size=2B, tensor_dim0/1, tile_dim0/1, tensor_dim0_stride. 2D tile -> groups 2/3 zero.
__device__ __forceinline__ void tdm_load_tile_2d(
    unsigned int lds_addr, const _Float16* gptr,
    unsigned int tensor_d0, unsigned int tensor_d1,
    unsigned int tile_d0, unsigned int tile_d1, unsigned int stride_elems)
{
    unsigned long long ga = (unsigned long long)(uintptr_t)gptr;
    v4u g0;
    g0.x = 1u;                                     // count=1, user descriptor
    g0.y = lds_addr;                               // LDS byte address
    g0.z = (unsigned int)ga;                       // global_addr[31:0]
    g0.w = (unsigned int)(ga >> 32) | 0x80000000u; // global_addr[56:32] | type=2 ([127:126])
    v8i g1;
    g1[0] = (int)(1u << 16);                                            // data_size=1 -> 2 bytes
    g1[1] = (int)((tensor_d0 & 0xFFFFu) << 16);                         // tensor_dim0 lo @ [63:48]
    g1[2] = (int)((tensor_d0 >> 16) | ((tensor_d1 & 0xFFFFu) << 16));   // dim0 hi | dim1 lo
    g1[3] = (int)((tensor_d1 >> 16) | (tile_d0 << 16));                 // dim1 hi | tile_dim0
    g1[4] = (int)(tile_d1 & 0xFFFFu);                                   // tile_dim1 (tile_dim2=0)
    g1[5] = (int)stride_elems;                                          // tensor_dim0_stride lo32
    g1[6] = 0;
    g1[7] = 0;
    v4i z4 = (v4i){0, 0, 0, 0};
#if defined(__clang_major__) && (__clang_major__ >= 23)
    v8i z8 = (v8i){0, 0, 0, 0, 0, 0, 0, 0};
    __builtin_amdgcn_tensor_load_to_lds(g0, g1, z4, z4, z8, 0);
#else
    __builtin_amdgcn_tensor_load_to_lds(g0, g1, z4, z4, 0);
#endif
}

// ---------------------------------------------------------------- LayerNorm
__global__ __launch_bounds__(256) void ln_to_f16(
    const float* __restrict__ X, const float* __restrict__ W,
    const float* __restrict__ Bb, _Float16* __restrict__ Y, int rows)
{
    int wid  = (blockIdx.x * blockDim.x + threadIdx.x) >> 5;
    int lane = threadIdx.x & 31;
    if (wid >= rows) return;
    const float* x = X + (size_t)wid * DIM;
    float v[24];
    float s = 0.f;
    #pragma unroll
    for (int i = 0; i < 24; ++i) { v[i] = x[i*32 + lane]; s += v[i]; }
    #pragma unroll
    for (int m = 1; m < 32; m <<= 1) s += __shfl_xor(s, m, 32);
    float mean = s * (1.0f/768.0f);
    float q = 0.f;
    #pragma unroll
    for (int i = 0; i < 24; ++i) { float d = v[i]-mean; q += d*d; }
    #pragma unroll
    for (int m = 1; m < 32; m <<= 1) q += __shfl_xor(q, m, 32);
    float rinv = rsqrtf(q*(1.0f/768.0f) + 1e-5f);
    _Float16* y = Y + (size_t)wid * DIM;
    #pragma unroll
    for (int i = 0; i < 24; ++i) {
        int c = i*32 + lane;
        y[c] = (_Float16)((v[i]-mean)*rinv*W[c] + Bb[c]);
    }
}

// ---------------------------------------------------------------- weight f32[K][N] -> f16 W^T[N][K]
__global__ void cvt_w_t(const float* __restrict__ x, _Float16* __restrict__ y, int K, int N)
{
    int i = blockIdx.x * 256 + threadIdx.x;
    if (i < K * N) {
        int k = i / N, n = i % N;
        y[(size_t)n * K + k] = (_Float16)x[i];
    }
}

// ---------------------------------------------------------------- 2D RoPE
__global__ void rope_f16(const float* __restrict__ X, const int* __restrict__ pos,
                         _Float16* __restrict__ Y)
{
    int tid  = blockIdx.x * 256 + threadIdx.x;      // ((b*12+h)*2048+p)*32 + j
    int j    = tid & 31;
    int p    = (tid >> 5) & 2047;
    int bh   = tid >> 16;
    int b    = bh / NH;
    int half = j >> 4, i = j & 15;
    int d0   = half*32 + 2*i;
    float freq = __expf(-(float)(2*i) * (4.60517018598809136804f / 32.0f)); // 100^(-2i/32)
    float ang  = (float)pos[((size_t)(b*PSEQ + p))*2 + half] * freq;
    float sn, cs; __sincosf(ang, &sn, &cs);
    size_t idx = (size_t)bh * PSEQ * DH + (size_t)p * DH + d0;
    float xe = X[idx], xo = X[idx+1];
    Y[idx]   = (_Float16)(xe*cs - xo*sn);
    Y[idx+1] = (_Float16)(xe*sn + xo*cs);
}

// ---------------------------------------------------------------- WMMA GEMM (TDM-staged)
// C[M,N] = A[M,K](f16,row) x Wt[N,K](f16,row, pre-transposed) + bias.
// Block 256 = 8 waves; tile 128x128; wave = 32x64 = 2x4 accums; K step 32.
// TDM double-buffers both 128x32 tiles into LDS; s_wait_tensorcnt + barrier hand off.
// MODE 0: f32 head layout [B,H,P,64]   (Q/K pre-rope)
// MODE 1: f16 transposed head [B,H,64,P] (V)
// MODE 2: gelu -> f16 row-major        (MLP hidden)
// MODE 3: + residual -> f32 row-major  (wo / w2)
template<int MODE>
__global__ __launch_bounds__(256) void gemm_f16_wmma(
    const _Float16* __restrict__ A, const _Float16* __restrict__ Wt,
    const float* __restrict__ bias, const float* __restrict__ res,
    float* __restrict__ outF, _Float16* __restrict__ outH,
    int M, int N, int K)
{
    __shared__ _Float16 sA[2][128*32];   // [row][k]
    __shared__ _Float16 sBt[2][128*32];  // [n][k]
    const int t    = threadIdx.x;
    const int lane = t & 31;
    const int wave = t >> 5;
    const int wm = wave >> 1, wn = wave & 1;
    const int m0 = blockIdx.y * 128, n0 = blockIdx.x * 128;

    // defeat "never-stored shared" folding; TDM writes are invisible to alias analysis
    if (t == 0) { sA[0][0] = (_Float16)0; sA[1][0] = (_Float16)0;
                  sBt[0][0] = (_Float16)0; sBt[1][0] = (_Float16)0; }

    v8f acc[2][4];
    #pragma unroll
    for (int i = 0; i < 2; ++i)
        #pragma unroll
        for (int j = 0; j < 4; ++j) acc[i][j] = (v8f){};

    const int nk = K >> 5;
    if (wave == 0) {  // prologue: stage tile 0 into buffer 0
        tdm_load_tile_2d((unsigned int)(uintptr_t)&sA[0][0],  A  + (size_t)m0 * K, K, M, 32, 128, K);
        tdm_load_tile_2d((unsigned int)(uintptr_t)&sBt[0][0], Wt + (size_t)n0 * K, K, N, 32, 128, K);
    }

    for (int kt = 0; kt < nk; ++kt) {
        const int buf = kt & 1;
        if (wave == 0) {
            if (kt + 1 < nk) {   // stage next tile into other buffer, keep 2 newest in flight
                const int k1 = (kt + 1) << 5;
                tdm_load_tile_2d((unsigned int)(uintptr_t)&sA[buf ^ 1][0],
                                 A + (size_t)m0 * K + k1, K, M, 32, 128, K);
                tdm_load_tile_2d((unsigned int)(uintptr_t)&sBt[buf ^ 1][0],
                                 Wt + (size_t)n0 * K + k1, K, N, 32, 128, K);
                __builtin_amdgcn_s_wait_tensorcnt(2);   // current tile's pair retired
            } else {
                __builtin_amdgcn_s_wait_tensorcnt(0);
            }
        }
        __syncthreads();

        v16h af[2], bf[4];
        #pragma unroll
        for (int im = 0; im < 2; ++im) {
            // 16-bit A 16x32 layout: lane m = lane%16; halves k = (lane<16?0:16)+h (contiguous)
            const _Float16* p = &sA[buf][(wm*32 + im*16 + (lane & 15))*32 + ((lane < 16) ? 0 : 16)];
            v8h lo = *(const v8h*)p;
            v8h hi = *(const v8h*)(p + 8);
            af[im] = pack16(lo, hi);
        }
        #pragma unroll
        for (int in = 0; in < 4; ++in) {
            // B 32x16 layout: lane n = lane%16; halves k = (lane<16?0:16)+h (contiguous)
            const _Float16* p = &sBt[buf][(wn*64 + in*16 + (lane & 15))*32 + ((lane < 16) ? 0 : 16)];
            bf[in] = *(const v16h*)p;
        }
        #pragma unroll
        for (int im = 0; im < 2; ++im)
            #pragma unroll
            for (int in = 0; in < 4; ++in)
                acc[im][in] = wmma_f16(af[im], bf[in], acc[im][in]);
        __syncthreads();   // all waves done reading before this buffer is re-filled
    }

    const int hi8  = (lane < 16) ? 0 : 8;
    const int coll = lane & 15;
    #pragma unroll
    for (int im = 0; im < 2; ++im)
    #pragma unroll
    for (int in = 0; in < 4; ++in)
    #pragma unroll
    for (int r = 0; r < 8; ++r) {
        int row = m0 + wm*32 + im*16 + r + hi8;
        int col = n0 + wn*64 + in*16 + coll;
        float v = acc[im][in][r] + bias[col];
        if (MODE == 0) {
            int b = row >> 11, p = row & 2047;
            int h = col >> 6,  d = col & 63;
            outF[((size_t)(b*NH + h)*PSEQ + p)*DH + d] = v;
        } else if (MODE == 1) {
            int b = row >> 11, p = row & 2047;
            int h = col >> 6,  d = col & 63;
            outH[((size_t)(b*NH + h)*DH + d)*PSEQ + p] = (_Float16)v;
        } else if (MODE == 2) {
            float g = 0.5f * v * (1.0f + erff(v * 0.70710678118654752f));
            outH[(size_t)row * N + col] = (_Float16)g;
        } else {
            outF[(size_t)row * N + col] = v + res[(size_t)row * N + col];
        }
    }
}

// ---------------------------------------------------------------- Flash attention
// one wave per (b,h, 16-row q block); Dh=64 = 2 WMMA k-steps; 32 kv per iter.
// Q,K: f16 [B,H,P,64]; Vt: f16 [B,H,64,P]; O: f16 [B,P,768]
__global__ __launch_bounds__(256) void attn_fa(
    const _Float16* __restrict__ Q, const _Float16* __restrict__ Kh,
    const _Float16* __restrict__ Vt, _Float16* __restrict__ O)
{
    __shared__ _Float16 pbuf[8][16*32];   // per-wave P scratch (C-layout -> A-layout)
    const int lane = threadIdx.x & 31;
    const int w    = threadIdx.x >> 5;
    const int wid  = blockIdx.x * 8 + w;
    const int bh   = wid >> 7;            // 0..23
    const int q0   = (wid & 127) * 16;
    const _Float16* q  = Q  + (size_t)bh * PSEQ * DH;
    const _Float16* k  = Kh + (size_t)bh * PSEQ * DH;
    const _Float16* vt = Vt + (size_t)bh * DH * PSEQ;

    const int coll = lane & 15;
    const int hi8  = (lane < 16) ? 0 : 8;
    const int kb16 = (lane < 16) ? 0 : 16;

    v16h aq[2];
    #pragma unroll
    for (int ks = 0; ks < 2; ++ks) {
        const _Float16* p = q + (size_t)(q0 + coll) * DH + ks*32 + ((lane < 16) ? 0 : 16);
        v8h lo = *(const v8h*)p;
        v8h hi = *(const v8h*)(p + 8);
        aq[ks] = pack16(lo, hi);
    }

    v8f o[4];
    #pragma unroll
    for (int i = 0; i < 4; ++i) o[i] = (v8f){};
    float mr[8], lr[8];
    #pragma unroll
    for (int r = 0; r < 8; ++r) { mr[r] = -1e30f; lr[r] = 0.f; }

    for (int kv0 = 0; kv0 < PSEQ; kv0 += 32) {
        v8f s[2];
        #pragma unroll
        for (int tt = 0; tt < 2; ++tt) {
            // B frag of K^T: lane n = kv col; halves over d (contiguous in K row-major)
            const _Float16* kp = k + (size_t)(kv0 + tt*16 + coll) * DH;
            v16h b0 = *(const v16h*)(kp + kb16);
            v16h b1 = *(const v16h*)(kp + 32 + kb16);
            v8f c = (v8f){};
            c = wmma_f16(aq[0], b0, c);
            c = wmma_f16(aq[1], b1, c);
            s[tt] = c;
        }
        // online softmax; row r + hi8 lives in one 16-lane half -> xor 1/2/4/8 reduces it
        #pragma unroll
        for (int r = 0; r < 8; ++r) {
            float s0 = s[0][r] * 0.125f, s1 = s[1][r] * 0.125f;
            float mx = fmaxf(s0, s1);
            #pragma unroll
            for (int m = 1; m < 16; m <<= 1) mx = fmaxf(mx, __shfl_xor(mx, m, 32));
            float mn = fmaxf(mr[r], mx);
            float al = __expf(mr[r] - mn);
            float p0 = __expf(s0 - mn), p1 = __expf(s1 - mn);
            float ps = p0 + p1;
            #pragma unroll
            for (int m = 1; m < 16; m <<= 1) ps += __shfl_xor(ps, m, 32);
            lr[r] = lr[r] * al + ps;
            mr[r] = mn;
            #pragma unroll
            for (int dt = 0; dt < 4; ++dt) o[dt][r] *= al;
            pbuf[w][(r + hi8)*32 + coll]      = (_Float16)p0;
            pbuf[w][(r + hi8)*32 + 16 + coll] = (_Float16)p1;
        }
        asm volatile("s_wait_dscnt 0" ::: "memory");  // per-wave LDS RAW; DS in-order per wave
        v16h ap;
        {
            const _Float16* p = &pbuf[w][coll*32 + ((lane < 16) ? 0 : 16)];
            v8h lo = *(const v8h*)p;
            v8h hi = *(const v8h*)(p + 8);
            ap = pack16(lo, hi);
        }
        #pragma unroll
        for (int dt = 0; dt < 4; ++dt) {
            // B frag of V: lane n = d col; halves over kv (contiguous in Vt)
            const _Float16* vp = vt + (size_t)(dt*16 + coll) * PSEQ + kv0 + kb16;
            v16h bv = *(const v16h*)vp;
            o[dt] = wmma_f16(ap, bv, o[dt]);
        }
    }

    const int b = bh / NH, h = bh % NH;
    #pragma unroll
    for (int r = 0; r < 8; ++r) {
        float inv = 1.0f / lr[r];
        int row = b * PSEQ + q0 + r + hi8;
        #pragma unroll
        for (int dt = 0; dt < 4; ++dt)
            O[(size_t)row * DIM + h*DH + dt*16 + coll] = (_Float16)(o[dt][r] * inv);
    }
}

// ---------------------------------------------------------------- launch
extern "C" void kernel_launch(void* const* d_in, const int* in_sizes, int n_in,
                              void* d_out, int out_size, void* d_ws, size_t ws_size,
                              hipStream_t stream) {
    (void)in_sizes; (void)n_in; (void)out_size; (void)ws_size;
    const float* query  = (const float*)d_in[0];
    const float* kv     = (const float*)d_in[1];
    const int*   pos_q  = (const int*)d_in[2];
    const int*   pos_kv = (const int*)d_in[3];
    const float* ln_q_w = (const float*)d_in[4];
    const float* ln_q_b = (const float*)d_in[5];
    const float* ln_kv_w= (const float*)d_in[6];
    const float* ln_kv_b= (const float*)d_in[7];
    const float* wq = (const float*)d_in[8];   const float* bq = (const float*)d_in[9];
    const float* wk = (const float*)d_in[10];  const float* bk = (const float*)d_in[11];
    const float* wv = (const float*)d_in[12];  const float* bv = (const float*)d_in[13];
    const float* wo = (const float*)d_in[14];  const float* bo = (const float*)d_in[15];
    const float* ln_m_w = (const float*)d_in[16];
    const float* ln_m_b = (const float*)d_in[17];
    const float* w1 = (const float*)d_in[18];  const float* b1 = (const float*)d_in[19];
    const float* w2 = (const float*)d_in[20];  const float* b2 = (const float*)d_in[21];
    float* out = (float*)d_out;

    char* ws = (char*)d_ws;
    auto alloc = [&](size_t bytes) { char* p = ws; ws += (bytes + 255) & ~(size_t)255; return p; };
    _Float16* qn   = (_Float16*)alloc((size_t)ROWS * DIM * 2);
    _Float16* kvn  = (_Float16*)alloc((size_t)ROWS * DIM * 2);
    _Float16* wqh  = (_Float16*)alloc((size_t)DIM * DIM * 2);   // transposed [N][K]
    _Float16* wkh  = (_Float16*)alloc((size_t)DIM * DIM * 2);
    _Float16* wvh  = (_Float16*)alloc((size_t)DIM * DIM * 2);
    _Float16* woh  = (_Float16*)alloc((size_t)DIM * DIM * 2);
    _Float16* w1h  = (_Float16*)alloc((size_t)DIM * MLPH * 2);  // [3072][768]
    _Float16* w2h  = (_Float16*)alloc((size_t)MLPH * DIM * 2);  // [768][3072]
    float*    qhf  = (float*)   alloc((size_t)ROWS * DIM * 4);
    float*    khf  = (float*)   alloc((size_t)ROWS * DIM * 4);
    _Float16* qh16 = (_Float16*)alloc((size_t)ROWS * DIM * 2);
    _Float16* kh16 = (_Float16*)alloc((size_t)ROWS * DIM * 2);
    _Float16* vt16 = (_Float16*)alloc((size_t)ROWS * DIM * 2);
    _Float16* attn = (_Float16*)alloc((size_t)ROWS * DIM * 2);
    float*    xres = (float*)   alloc((size_t)ROWS * DIM * 4);
    _Float16* xln  = (_Float16*)alloc((size_t)ROWS * DIM * 2);
    _Float16* h1   = (_Float16*)alloc((size_t)ROWS * MLPH * 2);

    // weights -> f16, transposed to [N][K] for TDM-friendly B tiles
    int nw = DIM * DIM;
    cvt_w_t<<<(nw + 255)/256, 256, 0, stream>>>(wq, wqh, DIM, DIM);
    cvt_w_t<<<(nw + 255)/256, 256, 0, stream>>>(wk, wkh, DIM, DIM);
    cvt_w_t<<<(nw + 255)/256, 256, 0, stream>>>(wv, wvh, DIM, DIM);
    cvt_w_t<<<(nw + 255)/256, 256, 0, stream>>>(wo, woh, DIM, DIM);
    int nm = DIM * MLPH;
    cvt_w_t<<<(nm + 255)/256, 256, 0, stream>>>(w1, w1h, DIM, MLPH);
    cvt_w_t<<<(nm + 255)/256, 256, 0, stream>>>(w2, w2h, MLPH, DIM);

    // LayerNorms -> f16
    ln_to_f16<<<ROWS/8, 256, 0, stream>>>(query, ln_q_w, ln_q_b, qn, ROWS);
    ln_to_f16<<<ROWS/8, 256, 0, stream>>>(kv, ln_kv_w, ln_kv_b, kvn, ROWS);

    // QKV projections
    dim3 gp(DIM/128, ROWS/128);
    gemm_f16_wmma<0><<<gp, 256, 0, stream>>>(qn,  wqh, bq, nullptr, qhf, nullptr, ROWS, DIM, DIM);
    gemm_f16_wmma<0><<<gp, 256, 0, stream>>>(kvn, wkh, bk, nullptr, khf, nullptr, ROWS, DIM, DIM);
    gemm_f16_wmma<1><<<gp, 256, 0, stream>>>(kvn, wvh, bv, nullptr, nullptr, vt16, ROWS, DIM, DIM);

    // RoPE (f32 head layout -> f16 head layout)
    int nrope = BATCH * NH * PSEQ * 32;
    rope_f16<<<nrope/256, 256, 0, stream>>>(qhf, pos_q,  qh16);
    rope_f16<<<nrope/256, 256, 0, stream>>>(khf, pos_kv, kh16);

    // Flash attention: 3072 waves, 8 per block
    attn_fa<<<(BATCH*NH*(PSEQ/16))/8, 256, 0, stream>>>(qh16, kh16, vt16, attn);

    // x = query + attn @ wo + bo
    gemm_f16_wmma<3><<<gp, 256, 0, stream>>>(attn, woh, bo, query, xres, nullptr, ROWS, DIM, DIM);

    // MLP
    ln_to_f16<<<ROWS/8, 256, 0, stream>>>(xres, ln_m_w, ln_m_b, xln, ROWS);
    dim3 g1(MLPH/128, ROWS/128);
    gemm_f16_wmma<2><<<g1, 256, 0, stream>>>(xln, w1h, b1, nullptr, nullptr, h1, ROWS, MLPH, DIM);
    gemm_f16_wmma<3><<<gp, 256, 0, stream>>>(h1, w2h, b2, xres, out, nullptr, ROWS, DIM, MLPH);
}
